// MultiHeadAttentionLayer_65420941853360
// MI455X (gfx1250) — matrix operand order repaired and checked by reading, not compiled, not run on hardware
//
#include <hip/hip_runtime.h>
#include <hip/hip_bf16.h>

typedef __attribute__((ext_vector_type(16))) __bf16 v16bf;
typedef __attribute__((ext_vector_type(8)))  __bf16 v8bf;
typedef __attribute__((ext_vector_type(8)))  float  v8f;

#define N_NODES  50000
#define N_EDGES  400000
#define IN_DIM   256
#define TOT_COLS 512      // Q(128) | K(128) | V(256)
#define LDS_STRIDE 264    // 256 + 8 bf16 pad -> 132 dwords/row -> conflict-free ds_load_b128

// ---------------------------------------------------------------------------
// Convert Wq/Wk/Wv (f32, row-major [cols x 256]) into one bf16 buffer
// Wb[512 x 256] and concat biases into bcat[512].
// ---------------------------------------------------------------------------
__global__ void prep_weights(const float* __restrict__ Wq, const float* __restrict__ bq,
                             const float* __restrict__ Wk, const float* __restrict__ bk,
                             const float* __restrict__ Wv, const float* __restrict__ bv,
                             __bf16* __restrict__ Wb, float* __restrict__ bcat) {
    int row = blockIdx.x;          // 0..511
    int t   = threadIdx.x;         // 0..255
    const float* srcW;
    const float* srcB;
    int srow;
    if (row < 128)      { srcW = Wq; srcB = bq; srow = row; }
    else if (row < 256) { srcW = Wk; srcB = bk; srow = row - 128; }
    else                { srcW = Wv; srcB = bv; srow = row - 256; }
    Wb[(size_t)row * IN_DIM + t] = (__bf16)srcW[(size_t)srow * IN_DIM + t];
    if (t == 0) bcat[row] = srcB[srow];
}

__global__ void zero_f32(float* __restrict__ p, size_t n) {
    size_t i = (size_t)blockIdx.x * blockDim.x + threadIdx.x;
    size_t stride = (size_t)gridDim.x * blockDim.x;
    for (; i < n; i += stride) p[i] = 0.0f;
}

// ---------------------------------------------------------------------------
// Fused QKV GEMM: P[N x 512] = h[N x 256] @ Wb^T + bcat, bf16 WMMA, f32 acc.
// grid.x = 3125 row-tiles (16 rows each); 8 waves/block, wave w -> cols w*64..+63.
// A tile staged once per block in LDS (f32->bf16 converted once, reused 8x).
// ---------------------------------------------------------------------------
__global__ void __launch_bounds__(256)
gemm_qkv(const float* __restrict__ h, const __bf16* __restrict__ Wb,
         const float* __restrict__ bcat, float* __restrict__ P) {
    __shared__ __bf16 As[16 * LDS_STRIDE];   // 8.25 KB bf16 A tile (padded rows)

    const int tid     = threadIdx.x;
    const int rowBase = blockIdx.x << 4;     // 16 rows per block

    // ---- cooperative A-tile stage: thread t converts 16 consecutive floats ----
    {
        const int r = tid >> 4;              // 0..15
        const int s = (tid & 15) << 4;       // 0,16,...,240
        const float* srcp = h + (size_t)(rowBase + r) * IN_DIM + s;
        __bf16* dstp = As + r * LDS_STRIDE + s;
#pragma unroll
        for (int i = 0; i < 16; i += 4) {
            float4 f = *(const float4*)(srcp + i);
            dstp[i + 0] = (__bf16)f.x; dstp[i + 1] = (__bf16)f.y;
            dstp[i + 2] = (__bf16)f.z; dstp[i + 3] = (__bf16)f.w;
        }
    }
    __syncthreads();

    const int wave = tid >> 5;
    const int lane = tid & 31;
    const int colBase = wave << 6;           // 64 cols per wave

    const int M    = lane & 15;              // A-frag row
    const int kb   = (lane & 16) ? 16 : 0;   // A-frag K base within 32-K tile
    const int n    = lane & 15;              // B-frag column
    const int kOff = (lane & 16) ? 16 : 0;   // B-frag K base

    v8f c0 = {}, c1 = {}, c2 = {}, c3 = {};
    const __bf16* aBase = As + M * LDS_STRIDE + kb;

#pragma unroll
    for (int kk = 0; kk < IN_DIM; kk += 32) {
        // ---- A fragment: two conflict-free ds_load_b128 from staged tile ----
        union { v16bf v; v8bf half[2]; } a;
        a.half[0] = *(const v8bf*)(aBase + kk);        // K = kb .. kb+7
        a.half[1] = *(const v8bf*)(aBase + kk + 8);    // K = kb+8 .. kb+15

        // ---- B fragments: 32x16 bf16 each; column n = row n of Wb (contiguous K) ----
        const __bf16* bp = Wb + (size_t)(colBase + n) * IN_DIM + kk + kOff;
        v16bf b0 = *(const v16bf*)(bp);
        v16bf b1 = *(const v16bf*)(bp + 16 * IN_DIM);
        v16bf b2 = *(const v16bf*)(bp + 32 * IN_DIM);
        v16bf b3 = *(const v16bf*)(bp + 48 * IN_DIM);

        c0 = __builtin_amdgcn_wmma_f32_16x16x32_bf16(false, a.v, false, b0, (short)0, c0, false, false);
        c1 = __builtin_amdgcn_wmma_f32_16x16x32_bf16(false, a.v, false, b1, (short)0, c1, false, false);
        c2 = __builtin_amdgcn_wmma_f32_16x16x32_bf16(false, a.v, false, b2, (short)0, c2, false, false);
        c3 = __builtin_amdgcn_wmma_f32_16x16x32_bf16(false, a.v, false, b3, (short)0, c3, false, false);
    }

    // ---- store C + bias: VGPR i -> row = i + (lane>=16 ? 8 : 0), col = lane&15 ----
    const int mOff = (lane & 16) ? 8 : 0;
    v8f cs[4] = {c0, c1, c2, c3};
#pragma unroll
    for (int t = 0; t < 4; ++t) {
        const int col = colBase + (t << 4) + (lane & 15);
        const float bias = bcat[col];
#pragma unroll
        for (int i = 0; i < 8; ++i) {
            const int row = rowBase + mOff + i;
            P[(size_t)row * TOT_COLS + col] = cs[t][i] + bias;
        }
    }
}

// ---------------------------------------------------------------------------
// Edge stage: one wave per edge. Lane = dim within a 32-wide head.
// P row layout: [ Q(0..127) | K(128..255) | V(256..511) ]
// ---------------------------------------------------------------------------
__global__ void __launch_bounds__(256)
edge_scatter(const float* __restrict__ P, const int* __restrict__ src,
             const int* __restrict__ dst, const float* __restrict__ jac,
             float* __restrict__ out, float* __restrict__ z, int E) {
    const int wavesPerBlock = blockDim.x >> 5;
    const int e = blockIdx.x * wavesPerBlock + (threadIdx.x >> 5);
    if (e >= E) return;
    const int lane = threadIdx.x & 31;

    const int s = src[e];
    const int d = dst[e];
    const float* Ps = P + (size_t)s * TOT_COLS;
    const float* Pd = P + (size_t)d * TOT_COLS;

    float scores[8];
#pragma unroll
    for (int hh = 0; hh < 4; ++hh) {
        float prod = Ps[128 + hh * 32 + lane] * Pd[hh * 32 + lane];  // K[src]*Q[dst]
#pragma unroll
        for (int off = 16; off > 0; off >>= 1)
            prod += __shfl_xor(prod, off, 32);
        float sc = prod * 0.17677669529663687f;                      // 1/sqrt(32)
        sc = fminf(5.0f, fmaxf(-5.0f, sc));
        scores[hh] = expf(sc);
    }
    const float jv = jac[e];
    scores[4] = jv; scores[5] = jv; scores[6] = jv; scores[7] = jv;

    float* outd = out + (size_t)d * 256;
#pragma unroll
    for (int hh = 0; hh < 8; ++hh) {
        const float v = Ps[256 + hh * 32 + lane];
        atomicAdd(outd + hh * 32 + lane, v * scores[hh]);
    }
    if (lane == 0) {
        float* zd = z + (size_t)d * 8;
#pragma unroll
        for (int hh = 0; hh < 8; ++hh) atomicAdd(zd + hh, scores[hh]);
    }
}

__global__ void normalize(float* __restrict__ out, const float* __restrict__ z, size_t n) {
    size_t i = (size_t)blockIdx.x * blockDim.x + threadIdx.x;
    size_t stride = (size_t)gridDim.x * blockDim.x;
    for (; i < n; i += stride) out[i] = out[i] / z[i >> 5];  // i>>5 == node*8 + head
}

// ---------------------------------------------------------------------------
extern "C" void kernel_launch(void* const* d_in, const int* in_sizes, int n_in,
                              void* d_out, int out_size, void* d_ws, size_t ws_size,
                              hipStream_t stream) {
    (void)in_sizes; (void)n_in; (void)out_size; (void)ws_size;

    const float* h   = (const float*)d_in[0];
    const int*   src = (const int*)  d_in[1];
    const int*   dst = (const int*)  d_in[2];
    const float* jac = (const float*)d_in[3];
    const float* Wq  = (const float*)d_in[4];
    const float* bq  = (const float*)d_in[5];
    const float* Wk  = (const float*)d_in[6];
    const float* bk  = (const float*)d_in[7];
    const float* Wv  = (const float*)d_in[8];
    const float* bv  = (const float*)d_in[9];
    float* out = (float*)d_out;

    // workspace carve-up (all 256B aligned)
    char* w = (char*)d_ws;
    float*  P    = (float*)w;  w += (size_t)N_NODES * TOT_COLS * sizeof(float);   // 102.4 MB
    float*  z    = (float*)w;  w += (size_t)N_NODES * 8 * sizeof(float);          //   1.6 MB
    __bf16* Wb   = (__bf16*)w; w += (size_t)TOT_COLS * IN_DIM * sizeof(__bf16);   //   0.25 MB
    float*  bcat = (float*)w;  w += (size_t)TOT_COLS * sizeof(float);

    // 1) weights f32 -> bf16 (+ bias concat)
    prep_weights<<<TOT_COLS, IN_DIM, 0, stream>>>(Wq, bq, Wk, bk, Wv, bv, Wb, bcat);

    // 2) zero accumulators (out & z); harness poisons buffers with 0xAA
    zero_f32<<<2048, 256, 0, stream>>>(out, (size_t)N_NODES * 256);
    zero_f32<<<256, 256, 0, stream>>>(z, (size_t)N_NODES * 8);

    // 3) fused QKV projection via bf16 WMMA (50000/16 = 3125 exact row tiles)
    gemm_qkv<<<N_NODES / 16, 256, 0, stream>>>(h, Wb, bcat, P);

    // 4) per-edge attention scores + scatter-add (wave per edge, 8 edges/block)
    edge_scatter<<<(N_EDGES + 7) / 8, 256, 0, stream>>>(P, src, dst, jac, out, z, N_EDGES);

    // 5) out = wV / z
    normalize<<<2048, 256, 0, stream>>>(out, z, (size_t)N_NODES * 256);
}
